// MultiHeadAttention_9019431321973
// MI455X (gfx1250) — matrix operand order repaired and checked by reading, not compiled, not run on hardware
//
#include <hip/hip_runtime.h>
#include <hip/hip_bf16.h>

// ---------------- types ----------------
typedef __bf16 bf16_t;
typedef __attribute__((ext_vector_type(16))) __bf16 v16bf;
typedef __attribute__((ext_vector_type(8)))  __bf16 bf16x8;
typedef __attribute__((ext_vector_type(8)))  float  v8f;

#define E_  1024
#define S_  2048
#define B_  2
#define H_  16
#define D_  64
#define N3_ 3072
#define M_  4096   // B_*S_

// ---------------- CDNA5 async global->LDS copy (16B per lane) ----------------
// GV mode: 64-bit global address in a VGPR pair, per-lane LDS byte offset in a VGPR.
// Tracked by ASYNCcnt (s_wait_asynccnt), in-order completion per wave.
#define ASYNC_CP_B128(ldsoff_u32, gaddr_u64)                                   \
  asm volatile("global_load_async_to_lds_b128 %0, %1, off"                     \
               :: "v"(ldsoff_u32), "v"(gaddr_u64) : "memory")
#define WAIT_ASYNC0() asm volatile("s_wait_asynccnt 0x0" ::: "memory")
#define WAIT_ASYNC4() asm volatile("s_wait_asynccnt 0x4" ::: "memory")
#define WAIT_DS0()    asm volatile("s_wait_dscnt 0x0" ::: "memory")

__device__ __forceinline__ unsigned lds_off(const void* p) {
  // generic LDS pointer carries the LDS byte offset in its low 32 bits
  return (unsigned)(size_t)p;
}

// ---------------- WMMA helpers ----------------
__device__ __forceinline__ v8f wmma_bf16(v16bf a, v16bf b, v8f c) {
  // 8 args: (neg_a, A, neg_b, B, c_mod, C, reuse_a, reuse_b)
  return __builtin_amdgcn_wmma_f32_16x16x32_bf16(false, a, false, b, (short)0, c,
                                                 false, false);
}

// A-matrix 16x32 fragment: lane (r,half) holds K = 16*half+[0..15]
__device__ __forceinline__ v16bf ld_frag_a(const bf16_t* p, int half) {
  bf16x8 lo = *(const bf16x8*)(p + 16 * half);
  bf16x8 hi = *(const bf16x8*)(p + 16 * half + 8);
  v16bf a;
#pragma unroll
  for (int i = 0; i < 8; ++i) { a[i] = lo[i]; a[i + 8] = hi[i]; }
  return a;
}

// B-matrix 32x16 fragment: lane (c,half) holds K = 16*half+[0..15] for column c.
__device__ __forceinline__ v16bf ld_frag_b(const bf16_t* p) {
  bf16x8 lo = *(const bf16x8*)(p);
  bf16x8 hi = *(const bf16x8*)(p + 8);
  v16bf b;
#pragma unroll
  for (int i = 0; i < 8; ++i) { b[i] = lo[i]; b[i + 8] = hi[i]; }
  return b;
}

// ---------------- conversion kernels ----------------
__global__ void cvt_bf16(const float* __restrict__ src, bf16_t* __restrict__ dst, int n) {
  int i = (blockIdx.x * blockDim.x + threadIdx.x) * 4;
  if (i < n) {
    float4 v = *(const float4*)(src + i);
    dst[i + 0] = (bf16_t)v.x;
    dst[i + 1] = (bf16_t)v.y;
    dst[i + 2] = (bf16_t)v.z;
    dst[i + 3] = (bf16_t)v.w;
  }
}

// src: K x N fp32 (row-major), dst: N x K bf16 (row-major)
__global__ void transpose_cvt(const float* __restrict__ src, bf16_t* __restrict__ dst,
                              int K, int N) {
  __shared__ float tile[32][33];
  int n = blockIdx.x * 32 + threadIdx.x;
#pragma unroll
  for (int j = 0; j < 32; j += 8) {
    int k = blockIdx.y * 32 + threadIdx.y + j;
    tile[threadIdx.y + j][threadIdx.x] = src[(size_t)k * N + n];
  }
  __syncthreads();
  int k2 = blockIdx.y * 32 + threadIdx.x;
#pragma unroll
  for (int j = 0; j < 32; j += 8) {
    int n2 = blockIdx.x * 32 + threadIdx.y + j;
    dst[(size_t)n2 * K + k2] = (bf16_t)tile[threadIdx.x][threadIdx.y + j];
  }
}

// ---------------- GEMM core: 128(M) x 128(N) block tile, 8 waves of 32x64 ----------------
// Double-buffered async staging: DMA tile k+1 into LDS buffer ^1 while WMMAs
// consume tile k. s_wait_asynccnt 4 drains only the older buffer's 4 copies.
__device__ __forceinline__ void gemm_core(const bf16_t* __restrict__ A,
                                          const bf16_t* __restrict__ Bt,
                                          int K, int m0, int n0,
                                          bf16_t (*As)[128][48],
                                          bf16_t (*Bs)[128][48],
                                          v8f acc[8]) {
  const int t = threadIdx.x;
  const int lane = t & 31, r = lane & 15, half = lane >> 4;
  const int wid = t >> 5, wm = wid & 3, wn = wid >> 2;
  // staging map: 256 threads cover a 128x32 tile with two 16B chunks each
  const int lrow = t >> 1, lc = (t & 1) * 16;
  const bf16_t* gA = A + (size_t)(m0 + lrow) * K + lc;
  const bf16_t* gB = Bt + (size_t)(n0 + lrow) * K + lc;

  auto issue = [&](int buf, int k0) {
    ASYNC_CP_B128(lds_off(&As[buf][lrow][lc]),     (unsigned long long)(gA + k0));
    ASYNC_CP_B128(lds_off(&As[buf][lrow][lc + 8]), (unsigned long long)(gA + k0 + 8));
    ASYNC_CP_B128(lds_off(&Bs[buf][lrow][lc]),     (unsigned long long)(gB + k0));
    ASYNC_CP_B128(lds_off(&Bs[buf][lrow][lc + 8]), (unsigned long long)(gB + k0 + 8));
  };

  const int nk = K / 32;
  issue(0, 0);
  for (int i = 0; i < nk; ++i) {
    const int cur = i & 1;
    const bool more = (i + 1 < nk);
    if (more) {
      issue(cur ^ 1, (i + 1) * 32);   // prefetch next tile into other buffer
      WAIT_ASYNC4();                  // drain only current buffer's copies
    } else {
      WAIT_ASYNC0();
    }
    __syncthreads();

    // pre-load all fragments, then a dense WMMA burst
    v16bf a[2], b[4];
#pragma unroll
    for (int mi = 0; mi < 2; ++mi)
      a[mi] = ld_frag_a(&As[cur][wm * 32 + mi * 16 + r][0], half);
#pragma unroll
    for (int j = 0; j < 4; ++j)
      b[j] = ld_frag_b(&Bs[cur][wn * 64 + j * 16 + r][16 * half]);
#pragma unroll
    for (int mi = 0; mi < 2; ++mi)
#pragma unroll
      for (int j = 0; j < 4; ++j)
        acc[mi * 4 + j] = wmma_bf16(a[mi], b[j], acc[mi * 4 + j]);
    __syncthreads();  // reads of buf[cur] done before it is re-filled at i+1
  }
}

// ---------------- kernel 1: fused QKV projection + scatter ----------------
__global__ __launch_bounds__(256) void gemm_qkv(const bf16_t* __restrict__ X,
                                                const bf16_t* __restrict__ Wt,
                                                const float* __restrict__ bias,
                                                bf16_t* __restrict__ Q,
                                                bf16_t* __restrict__ Kd,
                                                bf16_t* __restrict__ Vt) {
  __shared__ alignas(16) bf16_t As[2][128][48];
  __shared__ alignas(16) bf16_t Bs[2][128][48];
  const int lane = threadIdx.x & 31, r = lane & 15, half = lane >> 4;
  const int wid = threadIdx.x >> 5, wm = wid & 3, wn = wid >> 2;
  const int m0 = blockIdx.y * 128, n0 = blockIdx.x * 128;

  v8f acc[8] = {};
  gemm_core(X, Wt, E_, m0, n0, As, Bs, acc);

#pragma unroll
  for (int mi = 0; mi < 2; ++mi)
#pragma unroll
    for (int j = 0; j < 4; ++j) {
      int n = n0 + wn * 64 + j * 16 + r;
      float bv = bias[n];
      int three = n >> 10, h = (n >> 6) & 15, d = n & 63;
#pragma unroll
      for (int i = 0; i < 8; ++i) {
        int m = m0 + wm * 32 + mi * 16 + i + 8 * half;
        float v = acc[mi * 4 + j][i] + bv;
        int b = m >> 11, s = m & 2047;
        if (three == 0) {
          // fold score scale 1/sqrt(D)=0.125 into Q
          Q[((size_t)(b * H_ + h) * S_ + s) * D_ + d] = (bf16_t)(v * 0.125f);
        } else if (three == 1) {
          Kd[((size_t)(b * H_ + h) * S_ + s) * D_ + d] = (bf16_t)v;
        } else {
          // V stored transposed: [b,h,d,s] so PV B-fragments are K-contiguous
          Vt[((size_t)(b * H_ + h) * D_ + d) * S_ + s] = (bf16_t)v;
        }
      }
    }
}

// ---------------- kernel 2: flash attention (online softmax) ----------------
// 8 waves/block, 16 query rows per wave, 64-key chunks, D=64.
__global__ __launch_bounds__(256) void attn(const bf16_t* __restrict__ Q,
                                            const bf16_t* __restrict__ Kd,
                                            const bf16_t* __restrict__ Vt,
                                            bf16_t* __restrict__ O) {
  __shared__ alignas(16) bf16_t P[8][16][72];  // per-wave P tile, padded rows
  const int lane = threadIdx.x & 31, r = lane & 15, half = lane >> 4;
  const int wid = threadIdx.x >> 5;
  const int bh = blockIdx.y, bb = bh >> 4, hh = bh & 15;
  const bf16_t* Qh = Q + (size_t)bh * S_ * D_;
  const bf16_t* Kh = Kd + (size_t)bh * S_ * D_;
  const bf16_t* Vh = Vt + (size_t)bh * D_ * S_;
  bf16_t* Oh = O + (size_t)bb * S_ * E_ + hh * D_;
  const int q0 = blockIdx.x * 128 + wid * 16;

  v16bf qf[2];
#pragma unroll
  for (int kk = 0; kk < 2; ++kk)
    qf[kk] = ld_frag_a(Qh + (size_t)(q0 + r) * D_ + 32 * kk, half);

  v8f o[4] = {};
  float mrun[8], lrun[8];
#pragma unroll
  for (int i = 0; i < 8; ++i) { mrun[i] = -3.0e38f; lrun[i] = 0.0f; }

  for (int key0 = 0; key0 < S_; key0 += 64) {
    // prefetch next key/value chunk (L2-resident stream) -> global_prefetch_b8
    if (key0 + 64 < S_) {
      __builtin_prefetch(Kh + (size_t)(key0 + 64 + r) * D_, 0, 0);
      __builtin_prefetch(Vh + (size_t)r * S_ + key0 + 64, 0, 0);
    }

    // ---- scores: 4 frags of (16q x 16k), K-dim = D = 64 -> 2 WMMAs each
    v8f sc[4] = {};
#pragma unroll
    for (int f = 0; f < 4; ++f)
#pragma unroll
      for (int kk = 0; kk < 2; ++kk) {
        v16bf kb = ld_frag_b(Kh + (size_t)(key0 + 16 * f + r) * D_ + 32 * kk + 16 * half);
        sc[f] = wmma_bf16(qf[kk], kb, sc[f]);
      }

    // ---- online softmax: rows live across 16-lane halves; xor masks 1,2,4,8 stay in-half
    float mnew[8], alpha[8];
#pragma unroll
    for (int i = 0; i < 8; ++i) {
      float lm = fmaxf(fmaxf(sc[0][i], sc[1][i]), fmaxf(sc[2][i], sc[3][i]));
#pragma unroll
      for (int msk = 1; msk < 16; msk <<= 1) lm = fmaxf(lm, __shfl_xor(lm, msk));
      mnew[i] = fmaxf(mrun[i], lm);
      alpha[i] = __expf(mrun[i] - mnew[i]);
    }
#pragma unroll
    for (int i = 0; i < 8; ++i) {
      float p0 = __expf(sc[0][i] - mnew[i]);
      float p1 = __expf(sc[1][i] - mnew[i]);
      float p2 = __expf(sc[2][i] - mnew[i]);
      float p3 = __expf(sc[3][i] - mnew[i]);
      int row = i + 8 * half;  // C-layout row for element i
      P[wid][row][r]      = (bf16_t)p0;
      P[wid][row][16 + r] = (bf16_t)p1;
      P[wid][row][32 + r] = (bf16_t)p2;
      P[wid][row][48 + r] = (bf16_t)p3;
      float rs = p0 + p1 + p2 + p3;
#pragma unroll
      for (int msk = 1; msk < 16; msk <<= 1) rs += __shfl_xor(rs, msk);
      lrun[i] = alpha[i] * lrun[i] + rs;
      mrun[i] = mnew[i];
    }
#pragma unroll
    for (int of = 0; of < 4; ++of)
#pragma unroll
      for (int i = 0; i < 8; ++i) o[of][i] *= alpha[i];

    // P tile is wave-private: a dscnt drain is enough (no workgroup barrier)
    WAIT_DS0();

    // ---- P(16x64) @ V: A-frags from LDS, B-frags from transposed V
    v16bf pa[2];
#pragma unroll
    for (int kk = 0; kk < 2; ++kk)
      pa[kk] = ld_frag_a(&P[wid][r][32 * kk], half);
#pragma unroll
    for (int of = 0; of < 4; ++of)
#pragma unroll
      for (int kk = 0; kk < 2; ++kk) {
        v16bf vb = ld_frag_b(Vh + (size_t)(16 * of + r) * S_ + key0 + 32 * kk + 16 * half);
        o[of] = wmma_bf16(pa[kk], vb, o[of]);
      }
  }

#pragma unroll
  for (int of = 0; of < 4; ++of)
#pragma unroll
    for (int i = 0; i < 8; ++i) {
      int row = q0 + i + 8 * half;
      Oh[(size_t)row * E_ + 16 * of + r] = (bf16_t)(o[of][i] / lrun[i]);
    }
}

// ---------------- kernel 3: output projection ----------------
__global__ __launch_bounds__(256) void gemm_out(const bf16_t* __restrict__ A,
                                                const bf16_t* __restrict__ Wt,
                                                const float* __restrict__ bias,
                                                float* __restrict__ out) {
  __shared__ alignas(16) bf16_t As[2][128][48];
  __shared__ alignas(16) bf16_t Bs[2][128][48];
  const int lane = threadIdx.x & 31, r = lane & 15, half = lane >> 4;
  const int wid = threadIdx.x >> 5, wm = wid & 3, wn = wid >> 2;
  const int m0 = blockIdx.y * 128, n0 = blockIdx.x * 128;

  v8f acc[8] = {};
  gemm_core(A, Wt, E_, m0, n0, As, Bs, acc);

#pragma unroll
  for (int mi = 0; mi < 2; ++mi)
#pragma unroll
    for (int j = 0; j < 4; ++j) {
      int n = n0 + wn * 64 + j * 16 + r;
      float bv = bias[n];
#pragma unroll
      for (int i = 0; i < 8; ++i) {
        int m = m0 + wm * 32 + mi * 16 + i + 8 * half;
        out[(size_t)m * E_ + n] = acc[mi * 4 + j][i] + bv;
      }
    }
}

// ---------------- launch ----------------
extern "C" void kernel_launch(void* const* d_in, const int* in_sizes, int n_in,
                              void* d_out, int out_size, void* d_ws, size_t ws_size,
                              hipStream_t stream) {
  const float* query = (const float*)d_in[0];
  const float* Wqkv  = (const float*)d_in[1];
  const float* bqkv  = (const float*)d_in[2];
  const float* Wout  = (const float*)d_in[3];
  const float* bout  = (const float*)d_in[4];
  float* out = (float*)d_out;

  char* ws = (char*)d_ws;
  size_t off = 0;
  bf16_t* Xbf   = (bf16_t*)(ws + off); off += (size_t)M_ * E_ * 2;   // 8 MB
  bf16_t* WqkvT = (bf16_t*)(ws + off); off += (size_t)N3_ * E_ * 2;  // 6 MB
  bf16_t* WoutT = (bf16_t*)(ws + off); off += (size_t)E_ * E_ * 2;   // 2 MB
  bf16_t* Qb    = (bf16_t*)(ws + off); off += (size_t)M_ * E_ * 2;   // 8 MB
  bf16_t* Kb    = (bf16_t*)(ws + off); off += (size_t)M_ * E_ * 2;   // 8 MB
  bf16_t* Vtb   = (bf16_t*)(ws + off); off += (size_t)M_ * E_ * 2;   // 8 MB
  bf16_t* Obf   = (bf16_t*)(ws + off); off += (size_t)M_ * E_ * 2;   // 8 MB

  // 1) precision conversion / weight transposition (one-time L2 fill)
  cvt_bf16<<<(M_ * E_ / 4 + 255) / 256, 256, 0, stream>>>(query, Xbf, M_ * E_);
  transpose_cvt<<<dim3(N3_ / 32, E_ / 32), dim3(32, 8), 0, stream>>>(Wqkv, WqkvT, E_, N3_);
  transpose_cvt<<<dim3(E_ / 32, E_ / 32), dim3(32, 8), 0, stream>>>(Wout, WoutT, E_, E_);

  // 2) fused QKV projection (scatters Q,K [b,h,s,d] and V transposed [b,h,d,s])
  gemm_qkv<<<dim3(N3_ / 128, M_ / 128), 256, 0, stream>>>(Xbf, WqkvT, bqkv, Qb, Kb, Vtb);

  // 3) flash attention: grid = (query tiles, B*H)
  attn<<<dim3(S_ / 128, B_ * H_), 256, 0, stream>>>(Qb, Kb, Vtb, Obf);

  // 4) output projection (fp32 result + bias)
  gemm_out<<<dim3(E_ / 128, M_ / 128), 256, 0, stream>>>(Obf, WoutT, bout, out);
}
